// DeformableConv2d_23785528885537
// MI455X (gfx1250) — matrix operand order repaired and checked by reading, not compiled, not run on hardware
//
#include <hip/hip_runtime.h>

// ---------------------------------------------------------------------------
// Deformable Conv2d forward for gfx1250 (MI455X), wave32, WMMA bf16 path.
// B=8, CIN=COUT=64, H=W=112, K=3, stride=1, pad=1, dil=1.
// GEMM view: out[m, p] = sum_ck Wmat[m, ck] * samp[ck, p],  ck = c*9 + tap.
// ---------------------------------------------------------------------------

#define HH 112
#define WW 112
#define HW (HH * WW)          // 12544
#define BATCH 8
#define CIN 64
#define COUT 64
#define CK 576                // CIN * 9
#define KSTEPS 18             // CK / 32
#define TN 32                 // positions per workgroup
#define TILES_PER_B (HW / TN) // 392
#define NBLOCKS (BATCH * TILES_PER_B) // 3136
#define PITCH 584             // TN-tile LDS row pitch in bf16 (576 + 8 pad)

typedef __attribute__((ext_vector_type(16))) __bf16          bf16x16;
typedef __attribute__((ext_vector_type(8)))  float           f32x8;
typedef __attribute__((ext_vector_type(8)))  unsigned short  u16x8;

union FragU { bf16x16 bf; u16x8 us[2]; };

__device__ __forceinline__ unsigned short f2bf(float f) {
    unsigned int u = __float_as_uint(f);
    return (unsigned short)((u + 0x7FFFu + ((u >> 16) & 1u)) >> 16); // RNE
}

// A-fragment element -> K offset within a 32-wide K step (contiguous order,
// matching the B reads): lanes 0-15 hold K 0..15, lanes 16-31 hold K 16..31.
__device__ __forceinline__ int a_kofs(int lane, int e) {
    return ((lane < 16) ? 0 : 16) + e;
}

// ---------------------------------------------------------------------------
// Kernel 0: swizzle f32 weights into bf16 WMMA A-fragment order.
//   deform frags: [kstep 18][mtile 4][lane 32][e 16]  -> 36864 ushorts
//   offset frags: [kstep 18][mtile 2][lane 32][e 16]  -> 18432 ushorts (M padded to 32)
// ---------------------------------------------------------------------------
__global__ void prep_frags(const float* __restrict__ dw,   // [64][576]
                           const float* __restrict__ ow,   // [18][576]
                           unsigned short* __restrict__ wsD,
                           unsigned short* __restrict__ wsO) {
    int i = blockIdx.x * 256 + threadIdx.x;
    if (i < 36864) {
        int e = i & 15, lane = (i >> 4) & 31, mt = (i >> 9) & 3, ks = i >> 11;
        int m = mt * 16 + (lane & 15);
        int K = ks * 32 + a_kofs(lane, e);
        wsD[i] = f2bf(dw[m * CK + K]);
    } else if (i < 55296) {
        int j = i - 36864;
        int e = j & 15, lane = (j >> 4) & 31, mt = (j >> 9) & 1, ks = j >> 10;
        int m = mt * 16 + (lane & 15);
        int K = ks * 32 + a_kofs(lane, e);
        wsO[j] = (m < 18) ? f2bf(ow[m * CK + K]) : (unsigned short)0;
    }
}

// ---------------------------------------------------------------------------
// Kernel 1: offset conv (3x3, pad 1) as WMMA GEMM: [32pad x 576] x [576 x 32].
// 128 threads = 4 waves; each wave owns one 16x16 tile (2 mtiles x 2 ntiles).
// ---------------------------------------------------------------------------
__global__ void __launch_bounds__(128)
offset_conv(const float* __restrict__ x,
            const unsigned short* __restrict__ wfrag,  // prepped offset_w frags
            const float* __restrict__ obias,           // [18]
            float* __restrict__ offs) {                // [8][18][112][112]
    __shared__ __align__(16) unsigned short sSamp[TN * PITCH]; // 36.5 KB
    int t = threadIdx.x;
    int wg = blockIdx.x;
    int b = wg / TILES_PER_B;
    int ipBase = (wg - b * TILES_PER_B) * TN;
    const float* xb = x + (size_t)b * (CIN * HW);

    // ---- build im2col B-tile (nearest taps, zero pad) ----
    for (int it = 0; it < (TN * CK) / 128; ++it) {       // 144 iterations
        int v = it * 128 + t;
        int p = v / CK, ck = v - p * CK;
        int c = ck / 9, tap = ck - c * 9;
        int ki = tap / 3, kj = tap - ki * 3;
        int ip = ipBase + p;
        int ho = ip / WW, wo = ip - ho * WW;
        int y = ho - 1 + ki, xw = wo - 1 + kj;
        float val = (y >= 0 && y < HH && xw >= 0 && xw < WW)
                        ? xb[c * HW + y * WW + xw] : 0.0f;
        sSamp[p * PITCH + ck] = f2bf(val);
    }
    __syncthreads();

    // ---- WMMA: one 16x16 tile per wave ----
    int lane = t & 31, wv = t >> 5;
    int mtile = wv >> 1, ntile = wv & 1;
    f32x8 acc = {};
    const u16x8* wf = (const u16x8*)wfrag;
    for (int ks = 0; ks < KSTEPS; ++ks) {
        FragU a, bb;
        int abase = (((ks * 2 + mtile) * 32) + lane) * 2; // u16x8 units
        a.us[0] = wf[abase];
        a.us[1] = wf[abase + 1];
        const u16x8* bp = (const u16x8*)&sSamp[(ntile * 16 + (lane & 15)) * PITCH
                                               + ks * 32 + ((lane < 16) ? 0 : 16)];
        bb.us[0] = bp[0];
        bb.us[1] = bp[1];
        acc = __builtin_amdgcn_wmma_f32_16x16x32_bf16(
                  false, a.bf, false, bb.bf, (short)0, acc, false, false);
    }

    // ---- store rows m < 18 ----
    int pcol = ipBase + ntile * 16 + (lane & 15);
    int rowAdd = (lane < 16) ? 0 : 8;
#pragma unroll
    for (int r = 0; r < 8; ++r) {
        int m = mtile * 16 + rowAdd + r;
        if (m < 18)
            offs[(b * 18 + m) * HW + pcol] = acc[r] + obias[m];
    }
}

// ---------------------------------------------------------------------------
// Kernel 2: bilinear gather + main WMMA GEMM: [64 x 576] x [576 x 32].
// 256 threads = 8 waves; each wave owns one 16x16 tile (4 mtiles x 2 ntiles).
// ---------------------------------------------------------------------------
__global__ void __launch_bounds__(256)
deform_main(const float* __restrict__ x,
            const unsigned short* __restrict__ wfrag,  // prepped deform_w frags
            const float* __restrict__ dbias,           // [64]
            const float* __restrict__ offs,            // from offset_conv
            float* __restrict__ out) {                 // [8][64][112][112]
    __shared__ __align__(16) unsigned short sSamp[TN * PITCH]; // 36.5 KB
    __shared__ int4   sIdx[TN * 9];                             // 4.5 KB
    __shared__ float4 sWt[TN * 9];                              // 4.5 KB
    int t = threadIdx.x;
    int wg = blockIdx.x;
    int b = wg / TILES_PER_B;
    int ipBase = (wg - b * TILES_PER_B) * TN;
    const float* xb = x + (size_t)b * (CIN * HW);

    // ---- stage 1: per (position, tap) bilinear coords & validity-folded weights
    for (int e = t; e < TN * 9; e += 256) {
        int p = e / 9, tap = e - p * 9;
        int ip = ipBase + p;
        int ho = ip / WW, wo = ip - ho * WW;
        int ki = tap / 3, kj = tap - ki * 3;
        const float* op = offs + ((b * 18 + 2 * tap) * HW + ip);
        float dy = op[0];
        float dx = op[HW];
        float py = dy + (float)(ho - 1 + ki);
        float px = dx + (float)(wo - 1 + kj);
        float y0f = floorf(py), x0f = floorf(px);
        int y0 = (int)y0f, x0 = (int)x0f;
        int y1 = y0 + 1, x1 = x0 + 1;
        float wy1 = py - y0f, wy0 = 1.0f - wy1;
        float wx1 = px - x0f, wx0 = 1.0f - wx1;
        bool vy0 = (y0 >= 0) && (y0 < HH), vy1 = (y1 >= 0) && (y1 < HH);
        bool vx0 = (x0 >= 0) && (x0 < WW), vx1 = (x1 >= 0) && (x1 < WW);
        int cy0 = min(max(y0, 0), HH - 1), cy1 = min(max(y1, 0), HH - 1);
        int cx0 = min(max(x0, 0), WW - 1), cx1 = min(max(x1, 0), WW - 1);
        sIdx[e] = make_int4(cy0 * WW + cx0, cy0 * WW + cx1,
                            cy1 * WW + cx0, cy1 * WW + cx1);
        sWt[e]  = make_float4(vy0 && vx0 ? wy0 * wx0 : 0.0f,
                              vy0 && vx1 ? wy0 * wx1 : 0.0f,
                              vy1 && vx0 ? wy1 * wx0 : 0.0f,
                              vy1 && vx1 ? wy1 * wx1 : 0.0f);
    }
    __syncthreads();

    // ---- stage 2: gather + lerp -> bf16 B-tile in LDS ----
    for (int it = 0; it < (TN * CK) / 256; ++it) {       // 72 iterations
        int v = it * 256 + t;
        int p = v / CK, ck = v - p * CK;
        int c = ck / 9, tap = ck - c * 9;
        int e = p * 9 + tap;
        int4   id = sIdx[e];
        float4 w  = sWt[e];
        const float* xc = xb + c * HW;
        float s = w.x * xc[id.x] + w.y * xc[id.y]
                + w.z * xc[id.z] + w.w * xc[id.w];
        sSamp[p * PITCH + ck] = f2bf(s);
    }
    __syncthreads();

    // ---- stage 3: WMMA, one 16x16 tile per wave ----
    int lane = t & 31, wv = t >> 5;
    int mtile = wv >> 1, ntile = wv & 1;
    f32x8 acc = {};
    const u16x8* wf = (const u16x8*)wfrag;
    for (int ks = 0; ks < KSTEPS; ++ks) {
        FragU a, bb;
        int abase = (((ks * 4 + mtile) * 32) + lane) * 2; // u16x8 units
        a.us[0] = wf[abase];
        a.us[1] = wf[abase + 1];
        const u16x8* bp = (const u16x8*)&sSamp[(ntile * 16 + (lane & 15)) * PITCH
                                               + ks * 32 + ((lane < 16) ? 0 : 16)];
        bb.us[0] = bp[0];
        bb.us[1] = bp[1];
        acc = __builtin_amdgcn_wmma_f32_16x16x32_bf16(
                  false, a.bf, false, bb.bf, (short)0, acc, false, false);
    }

    // ---- epilogue: bias + coalesced f32 stores ----
    int pcol = ipBase + ntile * 16 + (lane & 15);
    int rowAdd = (lane < 16) ? 0 : 8;
#pragma unroll
    for (int r = 0; r < 8; ++r) {
        int m = mtile * 16 + rowAdd + r;
        out[((b * COUT + m) * HW) + pcol] = acc[r] + dbias[m];
    }
}

// ---------------------------------------------------------------------------
extern "C" void kernel_launch(void* const* d_in, const int* in_sizes, int n_in,
                              void* d_out, int out_size, void* d_ws, size_t ws_size,
                              hipStream_t stream) {
    (void)in_sizes; (void)n_in; (void)out_size; (void)ws_size;
    const float* x        = (const float*)d_in[0];  // [8][64][112][112]
    const float* offset_w = (const float*)d_in[1];  // [18][64][3][3]
    const float* offset_b = (const float*)d_in[2];  // [18]
    const float* deform_w = (const float*)d_in[3];  // [64][64][3][3]
    const float* deform_b = (const float*)d_in[4];  // [64]
    float* out = (float*)d_out;

    // workspace: [0,73728) deform frags | [73728,110592) offset frags |
    //            [110592, +7225344) offset-conv output f32
    unsigned short* wsD = (unsigned short*)d_ws;
    unsigned short* wsO = (unsigned short*)((char*)d_ws + 73728);
    float*          offs = (float*)((char*)d_ws + 110592);

    prep_frags <<<216,     256, 0, stream>>>(deform_w, offset_w, wsD, wsO);
    offset_conv<<<NBLOCKS, 128, 0, stream>>>(x, wsO, offset_b, offs);
    deform_main<<<NBLOCKS, 256, 0, stream>>>(x, wsD, deform_b, offs, out);
}
